// SigmaGate_37177236914768
// MI455X (gfx1250) — matrix operand order repaired and checked by reading, not compiled, not run on hardware
//
#include <hip/hip_runtime.h>

// ---- CDNA5 WMMA types ----
typedef __attribute__((ext_vector_type(16))) __bf16 v16bf;
typedef __attribute__((ext_vector_type(8)))  float  v8f;

#define DDIM    4096
#define NEXP    64
#define TILE_M  32                 // rows per workgroup (2 M-tiles)
#define CHUNK   256                // K staged per iteration
#define KSTEP   32                 // WMMA bf16 K
#define NKS     (CHUNK / KSTEP)    // 8 WMMA K-steps per chunk
#define WROWSTR (CHUNK + 8)        // LDS W row stride (bf16): 264*2B = 528B = 33*16 -> b128-aligned, bank-staggered
#define AHALF   (2 * NKS * 32 * 8) // elems per A half-plane array (8 bf16 per lane slot = 16B stride)

static __device__ __forceinline__ unsigned lds_off32(const void* p) {
    // generic (flat) pointer to __shared__: LDS aperture lives in addr[63:32],
    // low 32 bits carry the wave-relative LDS byte offset (ISA 10.2 aperture map)
    return (unsigned)(unsigned long long)p;
}

// GLOBAL_LOAD_ASYNC_TO_LDS_B128, GVS mode: mem = SADDR + VADDR(u32 bytes); tracked by ASYNCcnt
static __device__ __forceinline__ void async_b128(unsigned lds_byte, unsigned gbyte, const void* sbase) {
    asm volatile("global_load_async_to_lds_b128 %0, %1, %2"
                 :: "v"(lds_byte), "v"(gbyte), "s"(sbase) : "memory");
}

static __device__ __forceinline__ unsigned fbits(float f) { return __builtin_bit_cast(unsigned, f); }
static __device__ __forceinline__ float    bitsf(unsigned u) { return __builtin_bit_cast(float, u); }

// ---------------- prep: W [64,4096] f32 -> bf16 hi/lo planes in d_ws ----------------
__global__ __launch_bounds__(256) void conv_w_kernel(const float* __restrict__ w,
                                                     unsigned short* __restrict__ whi,
                                                     unsigned short* __restrict__ wlo) {
    int i = (blockIdx.x * 256 + threadIdx.x) * 4;            // 64*4096 = 262144 elems, 256x256x4
    float4 v = *(const float4*)&w[i];
    unsigned u0 = fbits(v.x), u1 = fbits(v.y), u2 = fbits(v.z), u3 = fbits(v.w);
    unsigned h01 = (u0 >> 16) | (u1 & 0xffff0000u);
    unsigned h23 = (u2 >> 16) | (u3 & 0xffff0000u);
    unsigned l0 = fbits(v.x - bitsf(u0 & 0xffff0000u));
    unsigned l1 = fbits(v.y - bitsf(u1 & 0xffff0000u));
    unsigned l2 = fbits(v.z - bitsf(u2 & 0xffff0000u));
    unsigned l3 = fbits(v.w - bitsf(u3 & 0xffff0000u));
    unsigned lo01 = (l0 >> 16) | (l1 & 0xffff0000u);
    unsigned lo23 = (l2 >> 16) | (l3 & 0xffff0000u);
    *(uint2*)&whi[i] = make_uint2(h01, h23);
    *(uint2*)&wlo[i] = make_uint2(lo01, lo23);
}

// ---------------- main: logits GEMM (WMMA bf16 3-term f32 emulation) + softmax + top-8 ----------------
__global__ __launch_bounds__(256) void sigma_gate_kernel(
    const float*          __restrict__ x,      // [T, 4096] f32
    const unsigned short* __restrict__ whi,    // [64, 4096] bf16 hi plane
    const unsigned short* __restrict__ wlo,    // [64, 4096] bf16 lo plane
    int*   __restrict__ idx_out,               // [T, 8]
    float* __restrict__ w_out)                 // [T, 8]
{
    // A planes split into K-half arrays: per-lane slot = 8 bf16 = 16B -> 4-dword lane
    // stride, only lanes l / l+16 alias a bank (2-way) instead of 8-way at 32B stride.
    __shared__ unsigned short a_hi0[AHALF];                  // 8 KB  (K in [0,16) of each kstep)
    __shared__ unsigned short a_hi1[AHALF];                  // 8 KB  (K in [16,32))
    __shared__ unsigned short a_lo0[AHALF];                  // 8 KB
    __shared__ unsigned short a_lo1[AHALF];                  // 8 KB
    __shared__ unsigned short w_hi_s[NEXP * WROWSTR];        // 33 KB (row-major, padded)
    __shared__ unsigned short w_lo_s[NEXP * WROWSTR];        // 33 KB
    __shared__ float          lg[TILE_M * NEXP];             // 8 KB logits

    const int tid  = threadIdx.x;
    const int wave = tid >> 5;          // 0..7
    const int lane = tid & 31;
    const int mt   = wave >> 2;         // M-tile 0/1
    const int nt   = wave & 3;          // N-tile 0..3 (16 experts each)
    const int g    = lane >> 4;         // K-half group
    const int m    = lane & 15;
    const int row0 = blockIdx.x * TILE_M;

    const unsigned whi_base = lds_off32(w_hi_s);
    const unsigned wlo_base = lds_off32(w_lo_s);

    // independent accumulators per emulation term -> WMMAs in a K-step pipeline freely
    v8f acc0 = {}, acc1 = {}, acc2 = {};

    for (int k0 = 0; k0 < DDIM; k0 += CHUNK) {
        __syncthreads();   // previous chunk fully consumed before overwrite

        // ---- async-stage W bf16 planes: 64 x 256 per plane, 512B per instruction ----
        #pragma unroll
        for (int j = 0; j < 8; ++j) {
            unsigned u  = (unsigned)(j * 256 + tid);          // b128 unit id (2048 per plane)
            unsigned er = u >> 5, cu = u & 31;                // expert row, col/8
            unsigned go = (er * DDIM + (unsigned)k0 + cu * 8u) * 2u;
            unsigned lo = (er * WROWSTR + cu * 8u) * 2u;
            async_b128(whi_base + lo, go, whi);
            async_b128(wlo_base + lo, go, wlo);
        }

        // ---- stage x: load f32, split to bf16 hi/lo, store pre-shuffled A fragments ----
        #pragma unroll
        for (int i = 0; i < 8; ++i) {
            int f = (i * 256 + tid) * 4;                      // flat elem of 32x256 tile
            int r = f >> 8, c = f & 255;
            float4 v = *(const float4*)&x[(size_t)(row0 + r) * DDIM + k0 + c];
            unsigned u0 = fbits(v.x), u1 = fbits(v.y), u2 = fbits(v.z), u3 = fbits(v.w);
            unsigned h01 = (u0 >> 16) | (u1 & 0xffff0000u);
            unsigned h23 = (u2 >> 16) | (u3 & 0xffff0000u);
            unsigned l0 = fbits(v.x - bitsf(u0 & 0xffff0000u));
            unsigned l1 = fbits(v.y - bitsf(u1 & 0xffff0000u));
            unsigned l2 = fbits(v.z - bitsf(u2 & 0xffff0000u));
            unsigned l3 = fbits(v.w - bitsf(u3 & 0xffff0000u));
            unsigned lo01 = (l0 >> 16) | (l1 & 0xffff0000u);
            unsigned lo23 = (l2 >> 16) | (l3 & 0xffff0000u);
            // A-fragment shuffle (ISA 7.12.2 16-bit A 16x32): lane = m + 16*((K>>3)&1),
            // half array h = K>>4, elem-in-half = K&7; a float4 (4-aligned K) stays in one 8-block
            int ks = c >> 5, kc = c & 31;
            int h  = (kc >> 4) & 1;
            int e0 = kc & 7;
            int ln = (r & 15) + (((kc >> 3) & 1) << 4);
            int slot = (((r >> 4) * NKS + ks) * 32 + ln) << 3;   // 8 elems per lane slot
            unsigned short* dh = h ? a_hi1 : a_hi0;
            unsigned short* dl = h ? a_lo1 : a_lo0;
            *(uint2*)&dh[slot + e0] = make_uint2(h01, h23);      // ds_store_b64
            *(uint2*)&dl[slot + e0] = make_uint2(lo01, lo23);
        }

        // hint next x chunk toward caches while we compute
        if (k0 + CHUNK < DDIM) {
            __builtin_prefetch(&x[(size_t)(row0 + (tid >> 3)) * DDIM + k0 + CHUNK + (tid & 7) * 32], 0, 1);
        }

        asm volatile("s_wait_asynccnt 0x0" ::: "memory");     // W planes landed in LDS
        __syncthreads();

        // ---- 8 WMMA K-steps: fragments are pure ds_load_b128, no VALU shuffle ----
        #pragma unroll
        for (int ks = 0; ks < NKS; ++ks) {
            const int slot = ((mt * NKS + ks) * 32 + lane) << 3;
            // B 32x16 bf16: lane n=m (expert nt*16+m), lanes 0-15 K=0..7,16..23, lanes 16-31 K=8..15,24..31
            const unsigned short* pb  = &w_hi_s[(nt * 16 + m) * WROWSTR + ks * KSTEP + 8 * g];
            const unsigned short* pbl = &w_lo_s[(nt * 16 + m) * WROWSTR + ks * KSTEP + 8 * g];
            v16bf Ah, Al, Bh, Bl;
            ((float4*)&Ah)[0] = *(const float4*)&a_hi0[slot];  ((float4*)&Ah)[1] = *(const float4*)&a_hi1[slot];
            ((float4*)&Al)[0] = *(const float4*)&a_lo0[slot];  ((float4*)&Al)[1] = *(const float4*)&a_lo1[slot];
            ((float4*)&Bh)[0] = *(const float4*)pb;            ((float4*)&Bh)[1] = *(const float4*)(pb  + 16);
            ((float4*)&Bl)[0] = *(const float4*)pbl;           ((float4*)&Bl)[1] = *(const float4*)(pbl + 16);
            // 3-term bf16 emulation of f32 GEMM (~2^-16 rel error), independent accumulators
            acc0 = __builtin_amdgcn_wmma_f32_16x16x32_bf16(false, Ah, false, Bh, (short)0, acc0, false, false);
            acc1 = __builtin_amdgcn_wmma_f32_16x16x32_bf16(false, Al, false, Bh, (short)0, acc1, false, false);
            acc2 = __builtin_amdgcn_wmma_f32_16x16x32_bf16(false, Ah, false, Bl, (short)0, acc2, false, false);
        }
    }

    v8f acc = acc0 + acc1 + acc2;

    // ---- spill logits (C layout: lanes 0-15 M=j, lanes 16-31 M=j+8; N=lane%16) ----
    __syncthreads();
    #pragma unroll
    for (int j = 0; j < 8; ++j) {
        lg[(mt * 16 + j + 8 * g) * NEXP + nt * 16 + m] = acc[j];
    }
    __syncthreads();

    // ---- softmax + top-8 + renormalize: one row per thread ----
    if (tid < TILE_M) {
        float* rowl = &lg[tid * NEXP];
        float mx = rowl[0];
        #pragma unroll 4
        for (int e = 1; e < NEXP; ++e) mx = fmaxf(mx, rowl[e]);
        float z = 0.0f;
        #pragma unroll 4
        for (int e = 0; e < NEXP; ++e) { float v = __expf(rowl[e] - mx); rowl[e] = v; z += v; }
        float inv = 1.0f / z;
        #pragma unroll 4
        for (int e = 0; e < NEXP; ++e) rowl[e] *= inv;

        unsigned long long used = 0;
        int   ids[8];
        float vals[8];
        float s = 0.0f;
        #pragma unroll
        for (int kq = 0; kq < 8; ++kq) {
            float best = -1.0f; int bi = 0;
            for (int e = 0; e < NEXP; ++e) {
                if ((used >> e) & 1ull) continue;
                float v = rowl[e];
                if (v > best) { best = v; bi = e; }   // strict '>' + ascending scan: first-index ties (jax)
            }
            used |= 1ull << bi;
            ids[kq] = bi; vals[kq] = best; s += best;
        }
        float dn = 1.0f / (s + 1e-20f);
        size_t base = (size_t)(row0 + tid) * 8;
        #pragma unroll
        for (int kq = 0; kq < 8; ++kq) {
            idx_out[base + kq] = ids[kq];
            w_out[base + kq]   = vals[kq] * dn;
        }
    }
}

extern "C" void kernel_launch(void* const* d_in, const int* in_sizes, int n_in,
                              void* d_out, int out_size, void* d_ws, size_t ws_size,
                              hipStream_t stream) {
    (void)n_in; (void)ws_size; (void)out_size;
    const float* x = (const float*)d_in[0];        // [T, 4096]
    const float* w = (const float*)d_in[1];        // [64, 4096]
    const int T = in_sizes[0] / DDIM;              // 16384

    // workspace: bf16 hi/lo planes of W (2 * 64*4096 * 2B = 1 MB)
    unsigned short* whi = (unsigned short*)d_ws;
    unsigned short* wlo = whi + (size_t)NEXP * DDIM;

    conv_w_kernel<<<dim3(256), dim3(256), 0, stream>>>(w, whi, wlo);

    int*   idx_out = (int*)d_out;
    float* wgt_out = (float*)d_out + (size_t)T * 8;
    sigma_gate_kernel<<<dim3(T / TILE_M), dim3(256), 0, stream>>>(x, whi, wlo, idx_out, wgt_out);
}
